// MPNN_75591424409724
// MI455X (gfx1250) — matrix operand order repaired and checked by reading, not compiled, not run on hardware
//
#include <hip/hip_runtime.h>
#include <hip/hip_bf16.h>
#include <stdint.h>

typedef __attribute__((ext_vector_type(16))) _Float16 v16h;
typedef __attribute__((ext_vector_type(8)))  _Float16 v8h;
typedef __attribute__((ext_vector_type(8)))  float    v8f;

#define NN 100000
#define DD 8
#define TT 3
#define AFD 70
#define BFD 6
#define HS 96            // padded fp16 h stride = 3 * 32
#define KC 3             // K chunks of 32 covering 70 (zero padded)
#define NTILES (NN/16)   // 6250 exactly
#define LSTR 104         // LDS stage row stride in halves (16B aligned, conflict-free b128 reads)

// fragment ids in packed-B buffer: layout [frag][lane(32)][dw(8)] uint32
#define F_W1(kc,nt) ((kc)*5 + (nt))          // 0..14   (U1)
#define F_W2(kc,nt) (15 + (kc)*5 + (nt))     // 15..29  (U2@V_w)
#define F_W3(nt)    (30 + (nt))              // 30..34  (U3@E_w, kc=0 only)
#define F_R(kc,nt)  (35 + (kc)*8 + (nt))     // 35..58  (R_w[:,:70]+R_w[:,70:])
#define NFRAG 59

union H16 { v16h v; v8h h[2]; uint4 q[2]; };

// ---------------- weight packing (exact B-fragment layout) ----------------

__device__ __forceinline__ float w_elem(int type, int ng, int k,
    const float* U_w, const float* V_w, const float* E_w, const float* R_w) {
  if (type == 0) {                       // W1 = U_w[:, 0:70]
    if (ng >= AFD || k >= AFD) return 0.f;
    return U_w[ng*146 + k];
  } else if (type == 1) {                // W2 = U_w[:,70:140] @ V_w
    if (ng >= AFD || k >= AFD) return 0.f;
    float s = 0.f;
    for (int j = 0; j < AFD; ++j) s += U_w[ng*146 + 70 + j] * V_w[j*AFD + k];
    return s;
  } else if (type == 2) {                // W3 = U_w[:,140:146] @ E_w
    if (ng >= AFD || k >= BFD) return 0.f;
    float s = 0.f;
    for (int j = 0; j < BFD; ++j) s += U_w[ng*146 + 140 + j] * E_w[j*BFD + k];
    return s;
  } else {                               // Rsum = R_w[:,:70] + R_w[:,70:140]
    if (ng >= 128 || k >= AFD) return 0.f;
    return R_w[ng*140 + k] + R_w[ng*140 + 70 + k];
  }
}

__global__ void pack_weights_kernel(const float* __restrict__ U_w,
                                    const float* __restrict__ V_w,
                                    const float* __restrict__ E_w,
                                    const float* __restrict__ R_w,
                                    uint32_t* __restrict__ bpack) {
  int t = blockIdx.x * blockDim.x + threadIdx.x;
  if (t >= NFRAG * 32) return;
  int frag = t >> 5, lane = t & 31;
  int n = lane & 15;
  int khalf = (lane >> 4) * 8;
  int type, kc, nt;
  if (frag < 15)      { type = 0; kc = frag / 5;        nt = frag % 5; }
  else if (frag < 30) { type = 1; kc = (frag - 15) / 5; nt = (frag - 15) % 5; }
  else if (frag < 35) { type = 2; kc = 0;               nt = frag - 30; }
  else                { type = 3; kc = (frag - 35) / 8; nt = (frag - 35) % 8; }
  int ng = nt * 16 + n;
  uint32_t* dst = bpack + ((size_t)frag * 32 + lane) * 8;
  for (int j = 0; j < 8; ++j) {
    int k0 = kc * 32 + khalf + 2 * j + (j >= 4 ? 8 : 0);
    float v0 = w_elem(type, ng, k0,     U_w, V_w, E_w, R_w);
    float v1 = w_elem(type, ng, k0 + 1, U_w, V_w, E_w, R_w);
    union { _Float16 h[2]; uint32_t u; } pk;
    pk.h[0] = (_Float16)v0; pk.h[1] = (_Float16)v1;
    dst[j] = pk.u;
  }
}

__global__ void fold_bias_kernel(const float* __restrict__ U_w,
                                 const float* __restrict__ U_b,
                                 const float* __restrict__ V_b,
                                 const float* __restrict__ E_b,
                                 float* __restrict__ biasF) {
  int i = blockIdx.x * blockDim.x + threadIdx.x;
  if (i >= 80) return;
  float b = 0.f;
  if (i < AFD) {
    b = U_b[i];
    for (int j = 0; j < AFD; ++j) b += U_w[i*146 + 70 + j] * V_b[j];
    for (int j = 0; j < BFD; ++j) b += U_w[i*146 + 140 + j] * E_b[j];
  }
  biasF[i] = b;   // biasF[70..79] = 0 so padded columns compute exact zeros
}

// ---------------- init / conversion kernels ----------------

__global__ void zero_h16_kernel(uint32_t* __restrict__ a, uint32_t* __restrict__ b, int n) {
  int i = blockIdx.x * blockDim.x + threadIdx.x;
  if (i < n) { a[i] = 0u; b[i] = 0u; }
}

__global__ void fill_h_kernel(const float* __restrict__ h, _Float16* __restrict__ h16, int n) {
  int i = blockIdx.x * blockDim.x + threadIdx.x;
  if (i < n) {
    int row = i / AFD, col = i % AFD;
    h16[(size_t)row * HS + col] = (_Float16)h[i];
  }
}

__global__ void fill_e_kernel(const float* __restrict__ e, _Float16* __restrict__ e16, int n) {
  int i = blockIdx.x * blockDim.x + threadIdx.x;
  if (i < n) {
    int nd = i >> 3, c = i & 7;  // padded [N*D][8]
    e16[i] = (c < BFD) ? (_Float16)e[nd * BFD + c] : (_Float16)0.f;
  }
}

__global__ void zero_feat_kernel(float* __restrict__ gfeat) {
  int i = threadIdx.x;
  if (i < 128) gfeat[i] = 0.f;
}

// ---------------- message-passing step: h' = relu(hc@W1^T + hc[g]@W2^T + e_d@W3^T + bias)

__global__ __launch_bounds__(256) void mp_step_kernel(
    const _Float16* __restrict__ hsrc, _Float16* __restrict__ hdst,
    const _Float16* __restrict__ e16, const int* __restrict__ g_idx,
    const uint32_t* __restrict__ bpack, const float* __restrict__ biasF, int d) {
  __shared__ _Float16 stage[8][16][LSTR];     // per-wave 16x80 result tile (padded stride)
  int gt = blockIdx.x * blockDim.x + threadIdx.x;
  int tile = gt >> 5;
  if (tile >= NTILES) return;                 // wave-uniform: EXEC stays all-ones
  int lane = threadIdx.x & 31;
  int wv = threadIdx.x >> 5;
  int m  = lane & 15;                         // A row / C column index
  int hs = lane >> 4;                         // half-wave K selector
  int node = tile * 16 + m;

  // A fragments: self h (3 K-chunks), gathered h (3), edge (1, K=6 zero-padded)
  H16 Ah[KC], Ag[KC], Ae;
  const _Float16* hp = hsrc + (size_t)node * HS + hs * 8;
  int gidx = g_idx[node * DD + d];
  const _Float16* gp = hsrc + (size_t)gidx * HS + hs * 8;
#pragma unroll
  for (int kc = 0; kc < KC; ++kc) {
    Ah[kc].h[0] = *(const v8h*)(hp + kc * 32);
    Ah[kc].h[1] = *(const v8h*)(hp + kc * 32 + 16);
    Ag[kc].h[0] = *(const v8h*)(gp + kc * 32);
    Ag[kc].h[1] = *(const v8h*)(gp + kc * 32 + 16);
  }
  v8h zv = {};
  v8h el = *(const v8h*)(e16 + (size_t)(node * DD + d) * 8);
  Ae.h[0] = (hs == 0) ? el : zv;              // K 0..7 only (K 6..31 are zero)
  Ae.h[1] = zv;

  const uint4* bp = (const uint4*)bpack;
#pragma unroll
  for (int nt = 0; nt < 5; ++nt) {
    v8f c = {};
#pragma unroll
    for (int kc = 0; kc < KC; ++kc) {
      H16 B1, B2;
      size_t f1 = (size_t)F_W1(kc, nt) * 32 + lane;
      size_t f2 = (size_t)F_W2(kc, nt) * 32 + lane;
      B1.q[0] = bp[f1*2+0]; B1.q[1] = bp[f1*2+1];
      B2.q[0] = bp[f2*2+0]; B2.q[1] = bp[f2*2+1];
      c = __builtin_amdgcn_wmma_f32_16x16x32_f16(false, Ah[kc].v, false, B1.v,
                                                 (short)0, c, false, false);
      c = __builtin_amdgcn_wmma_f32_16x16x32_f16(false, Ag[kc].v, false, B2.v,
                                                 (short)0, c, false, false);
    }
    {
      H16 B3;
      size_t f3 = (size_t)F_W3(nt) * 32 + lane;
      B3.q[0] = bp[f3*2+0]; B3.q[1] = bp[f3*2+1];
      c = __builtin_amdgcn_wmma_f32_16x16x32_f16(false, Ae.v, false, B3.v,
                                                 (short)0, c, false, false);
    }
    // epilogue: bias + relu, stage into LDS (padded cols 70..79 are exact zeros)
    int f = nt * 16 + m;                      // output feature (C column)
    float b = biasF[f];                       // biasF[70..79] == 0
    int r0 = hs * 8;                          // C rows j + 8*hs
#pragma unroll
    for (int j = 0; j < 8; ++j) {
      float v = c[j] + b;
      v = v > 0.f ? v : 0.f;
      stage[wv][r0 + j][f] = (_Float16)v;
    }
  }

  // coalesced write-back: 16 rows x 10 chunks of 8 halves = 5 b128 stores per lane
  // (LDS ops are in-order within a wave; same-wave RAW needs no barrier)
#pragma unroll
  for (int it = 0; it < 5; ++it) {
    int a   = it * 32 + lane;                 // 0..159
    int row = a & 15;
    int ch  = a >> 4;                         // 0..9 -> features ch*8..ch*8+7
    v8h val = *(const v8h*)&stage[wv][row][ch * 8];
    *(v8h*)(hdst + (size_t)(tile * 16 + row) * HS + ch * 8) = val;
  }
}

// ---------------- readout: feat = sum_n relu(h @ Rsum^T + R_b) ----------------

__global__ __launch_bounds__(256) void readout_kernel(
    const _Float16* __restrict__ hfin, const uint32_t* __restrict__ bpack,
    const float* __restrict__ R_b, float* __restrict__ gfeat) {
  __shared__ float sfeat[128];
  for (int i = threadIdx.x; i < 128; i += blockDim.x) sfeat[i] = 0.f;
  __syncthreads();

  int gt = blockIdx.x * blockDim.x + threadIdx.x;
  int tile = gt >> 5;
  int lane = threadIdx.x & 31;
  int m  = lane & 15;
  int hs = lane >> 4;
  if (tile < NTILES) {                        // wave-uniform
    int node = tile * 16 + m;
    H16 Ah[KC];
    const _Float16* hp = hfin + (size_t)node * HS + hs * 8;
#pragma unroll
    for (int kc = 0; kc < KC; ++kc) {
      Ah[kc].h[0] = *(const v8h*)(hp + kc * 32);
      Ah[kc].h[1] = *(const v8h*)(hp + kc * 32 + 16);
    }
    const uint4* bp = (const uint4*)bpack;
#pragma unroll
    for (int nt = 0; nt < 8; ++nt) {
      v8f c = {};
#pragma unroll
      for (int kc = 0; kc < KC; ++kc) {
        H16 B;
        size_t fr = (size_t)F_R(kc, nt) * 32 + lane;
        B.q[0] = bp[fr*2+0]; B.q[1] = bp[fr*2+1];
        c = __builtin_amdgcn_wmma_f32_16x16x32_f16(false, Ah[kc].v, false, B.v,
                                                   (short)0, c, false, false);
      }
      int f = nt * 16 + m;
      float rb = R_b[f];
      float s = 0.f;
#pragma unroll
      for (int j = 0; j < 8; ++j) {
        float v = c[j] + rb;
        s += v > 0.f ? v : 0.f;
      }
      atomicAdd(&sfeat[f], s);                // ds_add_f32, 16 writers per f
    }
  }
  __syncthreads();
  for (int i = threadIdx.x; i < 128; i += blockDim.x) atomicAdd(&gfeat[i], sfeat[i]);
}

// ---------------- tiny final MLP (single block) ----------------

__global__ __launch_bounds__(128) void mlp_kernel(
    const float* __restrict__ gfeat,
    const float* __restrict__ sh1_w, const float* __restrict__ sh1_b,
    const float* __restrict__ sh2_w, const float* __restrict__ sh2_b,
    const float* __restrict__ hid_w, const float* __restrict__ hid_b,
    const float* __restrict__ out_w, const float* __restrict__ out_b,
    float* __restrict__ out) {
  __shared__ float s1[128], s2[100], s3[100];
  int t = threadIdx.x;
  if (t < 128) {
    float a = sh1_b[t];
    for (int j = 0; j < 128; ++j) a += gfeat[j] * sh1_w[t*128 + j];
    s1[t] = a > 0.f ? a : 0.f;                // relu(relu(x)) == relu(x)
  }
  __syncthreads();
  if (t < 100) {
    float a = sh2_b[t];
    for (int j = 0; j < 128; ++j) a += s1[j] * sh2_w[t*128 + j];
    s2[t] = a;                                // "shared" (no relu here)
  }
  __syncthreads();
  if (t < 100) {
    float a = hid_b[t];
    for (int j = 0; j < 100; ++j) a += s2[j] * hid_w[t*100 + j];
    s3[t] = a > 0.f ? a : 0.f;                // hid; relu(hid)==hid afterwards
  }
  __syncthreads();
  if (t == 0) {
    float a = out_b[0];
    for (int j = 0; j < 100; ++j) a += s3[j] * out_w[j];
    out[0] = a; out[1] = a; out[2] = a;       // identical across NT=3 tasks
  }
}

// ---------------- launch ----------------

extern "C" void kernel_launch(void* const* d_in, const int* in_sizes, int n_in,
                              void* d_out, int out_size, void* d_ws, size_t ws_size,
                              hipStream_t stream) {
  const float* h     = (const float*)d_in[0];
  const float* e     = (const float*)d_in[1];
  const float* R_w   = (const float*)d_in[2];
  const float* R_b   = (const float*)d_in[3];
  const float* V_w   = (const float*)d_in[4];
  const float* V_b   = (const float*)d_in[5];
  const float* E_w   = (const float*)d_in[6];
  const float* E_b   = (const float*)d_in[7];
  const float* U_w   = (const float*)d_in[8];
  const float* U_b   = (const float*)d_in[9];
  const float* sh1_w = (const float*)d_in[10];
  const float* sh1_b = (const float*)d_in[11];
  const float* sh2_w = (const float*)d_in[12];
  const float* sh2_b = (const float*)d_in[13];
  const float* hid_w = (const float*)d_in[14];
  const float* hid_b = (const float*)d_in[15];
  const float* out_w = (const float*)d_in[16];
  const float* out_b = (const float*)d_in[17];
  const int*   g_idx = (const int*)d_in[18];
  float* out = (float*)d_out;
  (void)in_sizes; (void)n_in; (void)out_size; (void)ws_size;

  char* ws = (char*)d_ws;
  size_t off = 0;
  auto alloc = [&](size_t bytes) -> void* {
    void* p = ws + off;
    off = (off + bytes + 255) & ~(size_t)255;
    return p;
  };
  _Float16* h16a  = (_Float16*)alloc((size_t)NN * HS * 2);        // 19.2 MB
  _Float16* h16b  = (_Float16*)alloc((size_t)NN * HS * 2);        // 19.2 MB
  _Float16* e16   = (_Float16*)alloc((size_t)NN * DD * 8 * 2);    // 12.8 MB
  uint32_t* bpack = (uint32_t*)alloc((size_t)NFRAG * 32 * 8 * 4); // 60 KB
  float*    biasF = (float*)alloc(80 * 4);
  float*    gfeat = (float*)alloc(128 * 4);

  // --- prep: zero padded fp16 buffers, convert inputs, fold+pack weights ---
  {
    int n = NN * (HS / 2);  // dwords per h16 buffer
    zero_h16_kernel<<<(n + 255) / 256, 256, 0, stream>>>((uint32_t*)h16a, (uint32_t*)h16b, n);
  }
  {
    int n = NN * AFD;
    fill_h_kernel<<<(n + 255) / 256, 256, 0, stream>>>(h, h16a, n);
  }
  {
    int n = NN * DD * 8;
    fill_e_kernel<<<(n + 255) / 256, 256, 0, stream>>>(e, e16, n);
  }
  fold_bias_kernel<<<1, 128, 0, stream>>>(U_w, U_b, V_b, E_b, biasF);
  {
    int n = NFRAG * 32;
    pack_weights_kernel<<<(n + 255) / 256, 256, 0, stream>>>(U_w, V_w, E_w, R_w, bpack);
  }

  // --- 24 sequential message-passing sweeps (double-buffered) ---
  const int grid = (NTILES * 32 + 255) / 256;
  const _Float16* src = h16a;
  _Float16* dst = h16b;
  for (int t = 0; t < TT; ++t) {
    for (int d = 0; d < DD; ++d) {
      mp_step_kernel<<<grid, 256, 0, stream>>>(src, dst, e16, g_idx, bpack, biasF, d);
      const _Float16* tmp = src; src = dst; dst = (_Float16*)tmp;
    }
  }
  // after 24 swaps, `src` holds the final h (== h16a)

  // --- readout + MLP head ---
  zero_feat_kernel<<<1, 128, 0, stream>>>(gfeat);
  readout_kernel<<<grid, 256, 0, stream>>>(src, bpack, R_b, gfeat);
  mlp_kernel<<<1, 128, 0, stream>>>(gfeat, sh1_w, sh1_b, sh2_w, sh2_b,
                                    hid_w, hid_b, out_w, out_b, out);
}
